// SimpleSAE_42374147342790
// MI455X (gfx1250) — matrix operand order repaired and checked by reading, not compiled, not run on hardware
//
#include <hip/hip_runtime.h>
#include <hip/hip_bf16.h>

#define D_MODEL 768
#define D_SAE   16384
#define K_TOP   32
#define N_ROWS  8192

typedef __attribute__((ext_vector_type(16))) __bf16 bf16x16;
typedef __attribute__((ext_vector_type(8)))  __bf16 bf16x8;
typedef __attribute__((ext_vector_type(8)))  float  f32x8;

// ---------- helpers ----------

__device__ __forceinline__ unsigned short f32_to_bf16_rne(float f) {
    unsigned int u = __float_as_uint(f);
    unsigned int lsb = (u >> 16) & 1u;
    u += 0x7fffu + lsb;                       // round-to-nearest-even
    return (unsigned short)(u >> 16);
}
__device__ __forceinline__ float bf16_to_f32(unsigned short h) {
    return __uint_as_float(((unsigned int)h) << 16);
}

// A-fragment (16x32 bf16): lane holds row M=lane&15; K = sel8 + {0..7} and sel8 + {16..23}
// where sel8 = (lane>=16 ? 8 : 0). Two contiguous 16B chunks.
__device__ __forceinline__ bf16x16 load_afrag(const unsigned short* p) {
    bf16x8 c0 = *(const bf16x8*)(p);
    bf16x8 c1 = *(const bf16x8*)(p + 16);
    return __builtin_shufflevector(c0, c1, 0,1,2,3,4,5,6,7,8,9,10,11,12,13,14,15);
}
// Contiguous-K load: 16 bf16 from p in one 32B chunk; unused (B frags use load_afrag, table order).
__device__ __forceinline__ bf16x16 load_bfrag(const unsigned short* p) {
    return *(const bf16x16*)(p);
}

__device__ __forceinline__ f32x8 wmma_bf16(bf16x16 a, bf16x16 b, f32x8 c) {
    return __builtin_amdgcn_wmma_f32_16x16x32_bf16(false, a, false, b, (short)0, c, false, false);
}

// ---------- kernel 1: split fp32 -> bf16 hi + bf16 lo (residual) ----------

__global__ void sae_split_bf16(const float* __restrict__ in,
                               unsigned short* __restrict__ hi,
                               unsigned short* __restrict__ lo,
                               int count) {
    int stride = gridDim.x * blockDim.x;
    for (int i = blockIdx.x * blockDim.x + threadIdx.x; i < count; i += stride) {
        float f = in[i];
        unsigned short h = f32_to_bf16_rne(f);
        hi[i] = h;
        lo[i] = f32_to_bf16_rne(f - bf16_to_f32(h));
    }
}

// ---------- kernel 2: transpose W_dec [768,16384] -> W_decT [16384,768] ----------

__global__ void sae_transpose_wdec(const float* __restrict__ W_dec,
                                   float* __restrict__ W_decT) {
    __shared__ float tile[32][33];
    int sBase = blockIdx.x * 32;
    int dBase = blockIdx.y * 32;
    int tx = threadIdx.x;   // 0..31
    int ty = threadIdx.y;   // 0..7
#pragma unroll
    for (int i = 0; i < 4; ++i)
        tile[ty + i * 8][tx] = W_dec[(size_t)(dBase + ty + i * 8) * D_SAE + sBase + tx];
    __syncthreads();
#pragma unroll
    for (int i = 0; i < 4; ++i)
        W_decT[(size_t)(sBase + ty + i * 8) * D_MODEL + dBase + tx] = tile[tx][ty + i * 8];
}

// ---------- kernel 3: encoder GEMM via split-bf16 WMMA ----------
// C[8192,16384] ; workgroup tile 64(M) x 128(N), 8 waves of 32x32, frags straight
// from global (all operands L2-resident on MI455X's 192MB L2).

__global__ __launch_bounds__(256)
void sae_encode_wmma(const unsigned short* __restrict__ x_hi,
                     const unsigned short* __restrict__ x_lo,
                     const unsigned short* __restrict__ we_hi,
                     const unsigned short* __restrict__ we_lo,
                     const float* __restrict__ b_enc,
                     float* __restrict__ latents,
                     float* __restrict__ sparse) {
    const int tid  = threadIdx.x;
    const int lane = tid & 31;
    const int wave = tid >> 5;
    const int wn   = wave & 3;          // 0..3 -> N offset
    const int wm   = wave >> 2;         // 0..1 -> M offset

    const int tileN = blockIdx.x * 128 + wn * 32;
    const int tileM = blockIdx.y * 64  + wm * 32;

    const int l15    = lane & 15;
    const int hiHalf = lane >> 4;
    const int selA   = hiHalf * 8;      // A frag K shift for upper half-wave
    const int selB   = hiHalf * 8;      // B frag K shift for upper half-wave

    const unsigned short* xh0 = x_hi + (size_t)(tileM + l15)      * D_MODEL;
    const unsigned short* xh1 = x_hi + (size_t)(tileM + 16 + l15) * D_MODEL;
    const unsigned short* xl0 = x_lo + (size_t)(tileM + l15)      * D_MODEL;
    const unsigned short* xl1 = x_lo + (size_t)(tileM + 16 + l15) * D_MODEL;
    const unsigned short* wh0 = we_hi + (size_t)(tileN + l15)      * D_MODEL;
    const unsigned short* wh1 = we_hi + (size_t)(tileN + 16 + l15) * D_MODEL;
    const unsigned short* wl0 = we_lo + (size_t)(tileN + l15)      * D_MODEL;
    const unsigned short* wl1 = we_lo + (size_t)(tileN + 16 + l15) * D_MODEL;

    f32x8 acc00 = {}, acc01 = {}, acc10 = {}, acc11 = {};

    for (int k0 = 0; k0 < D_MODEL; k0 += 32) {
        const int ka = k0 + selA;
        const int kb = k0 + selB;

        bf16x16 aH0 = load_afrag(xh0 + ka);
        bf16x16 aH1 = load_afrag(xh1 + ka);
        bf16x16 aL0 = load_afrag(xl0 + ka);
        bf16x16 aL1 = load_afrag(xl1 + ka);
        bf16x16 bH0 = load_afrag(wh0 + kb);
        bf16x16 bH1 = load_afrag(wh1 + kb);
        bf16x16 bL0 = load_afrag(wl0 + kb);
        bf16x16 bL1 = load_afrag(wl1 + kb);

        // acc += hi*hi + hi*lo + lo*hi   (fp32-grade accuracy, bf16 WMMA rate)
        acc00 = wmma_bf16(aH0, bH0, acc00);
        acc00 = wmma_bf16(aH0, bL0, acc00);
        acc00 = wmma_bf16(aL0, bH0, acc00);

        acc01 = wmma_bf16(aH0, bH1, acc01);
        acc01 = wmma_bf16(aH0, bL1, acc01);
        acc01 = wmma_bf16(aL0, bH1, acc01);

        acc10 = wmma_bf16(aH1, bH0, acc10);
        acc10 = wmma_bf16(aH1, bL0, acc10);
        acc10 = wmma_bf16(aL1, bH0, acc10);

        acc11 = wmma_bf16(aH1, bH1, acc11);
        acc11 = wmma_bf16(aH1, bL1, acc11);
        acc11 = wmma_bf16(aL1, bH1, acc11);
    }

    // C/D 16x16 f32 layout: VGPR v -> row = v + 8*hiHalf, col = lane&15
    const int col0 = tileN + l15;
    const int col1 = tileN + 16 + l15;
    const float be0 = b_enc[col0];
    const float be1 = b_enc[col1];

#pragma unroll
    for (int v = 0; v < 8; ++v) {
        const int r0 = tileM + hiHalf * 8 + v;
        const int r1 = tileM + 16 + hiHalf * 8 + v;
        const size_t o00 = (size_t)r0 * D_SAE + col0;
        const size_t o01 = (size_t)r0 * D_SAE + col1;
        const size_t o10 = (size_t)r1 * D_SAE + col0;
        const size_t o11 = (size_t)r1 * D_SAE + col1;
        latents[o00] = acc00[v] + be0;
        latents[o01] = acc01[v] + be1;
        latents[o10] = acc10[v] + be0;
        latents[o11] = acc11[v] + be1;
        // zero the sparse-latents region in the same pass
        sparse[o00] = 0.0f;
        sparse[o01] = 0.0f;
        sparse[o10] = 0.0f;
        sparse[o11] = 0.0f;
    }
}

// ---------- kernel 4: per-row top-32 (iterative argmax in LDS) + scatter ----------

__global__ __launch_bounds__(256)
void sae_topk(const float* __restrict__ latents,
              float* __restrict__ sparse,
              float* __restrict__ tk_vals,
              int*   __restrict__ tk_idx) {
    extern __shared__ float smem[];
    float* sv = smem;                       // [16384] row copy
    float* rv = smem + D_SAE;               // [256] reduction vals
    int*   ri = (int*)(smem + D_SAE + 256); // [256] reduction idx

    const int n   = blockIdx.x;
    const int tid = threadIdx.x;
    const float* row = latents + (size_t)n * D_SAE;

    for (int j = tid; j < D_SAE; j += 256) sv[j] = row[j];
    __syncthreads();

    for (int it = 0; it < K_TOP; ++it) {
        float best = -3.4e38f;
        int   bi   = 0;
        for (int j = tid; j < D_SAE; j += 256) {
            float v = sv[j];
            if (v > best) { best = v; bi = j; }
        }
        rv[tid] = best; ri[tid] = bi;
        __syncthreads();
        for (int s = 128; s > 0; s >>= 1) {
            if (tid < s && rv[tid + s] > rv[tid]) { rv[tid] = rv[tid + s]; ri[tid] = ri[tid + s]; }
            __syncthreads();
        }
        if (tid == 0) {
            float v = rv[0]; int idx = ri[0];
            tk_vals[n * K_TOP + it] = v;
            tk_idx [n * K_TOP + it] = idx;
            sparse[(size_t)n * D_SAE + idx] = v;
            sv[idx] = -3.4e38f;
        }
        __syncthreads();
    }
}

// ---------- kernel 5: sparse decode via gathered rows of W_decT ----------

__global__ __launch_bounds__(256)
void sae_decode(const float* __restrict__ tk_vals,
                const int*   __restrict__ tk_idx,
                const float* __restrict__ WdT,     // [16384][768], L2-resident
                const float* __restrict__ b_dec,
                float* __restrict__ rec) {
    __shared__ float kv[K_TOP];
    __shared__ int   ki[K_TOP];
    const int n   = blockIdx.x;
    const int tid = threadIdx.x;
    if (tid < K_TOP) {
        kv[tid] = tk_vals[n * K_TOP + tid];
        ki[tid] = tk_idx [n * K_TOP + tid];
    }
    __syncthreads();
    float a0 = 0.f, a1 = 0.f, a2 = 0.f;
#pragma unroll
    for (int k = 0; k < K_TOP; ++k) {
        const float v = kv[k];
        const float* w = WdT + (size_t)ki[k] * D_MODEL;
        a0 += v * w[tid];
        a1 += v * w[tid + 256];
        a2 += v * w[tid + 512];
    }
    const size_t o = (size_t)n * D_MODEL;
    rec[o + tid]       = a0 + b_dec[tid];
    rec[o + tid + 256] = a1 + b_dec[tid + 256];
    rec[o + tid + 512] = a2 + b_dec[tid + 512];
}

// ---------- launch ----------

extern "C" void kernel_launch(void* const* d_in, const int* in_sizes, int n_in,
                              void* d_out, int out_size, void* d_ws, size_t ws_size,
                              hipStream_t stream) {
    (void)in_sizes; (void)n_in; (void)out_size; (void)ws_size;

    const float* x     = (const float*)d_in[0];
    const float* W_enc = (const float*)d_in[1];
    const float* b_enc = (const float*)d_in[2];
    const float* W_dec = (const float*)d_in[3];
    const float* b_dec = (const float*)d_in[4];

    // output layout: rec | sparse_latents | latents (flat, in return order)
    float* rec     = (float*)d_out;
    float* sparse  = rec + (size_t)N_ROWS * D_MODEL;
    float* latents = sparse + (size_t)N_ROWS * D_SAE;

    // workspace layout (~122 MB)
    char* ws = (char*)d_ws;
    unsigned short* x_hi  = (unsigned short*)ws;
    unsigned short* x_lo  = x_hi + (size_t)N_ROWS * D_MODEL;
    unsigned short* we_hi = x_lo + (size_t)N_ROWS * D_MODEL;
    unsigned short* we_lo = we_hi + (size_t)D_SAE * D_MODEL;
    float* WdT     = (float*)(we_lo + (size_t)D_SAE * D_MODEL);
    float* tk_vals = WdT + (size_t)D_SAE * D_MODEL;
    int*   tk_idx  = (int*)(tk_vals + (size_t)N_ROWS * K_TOP);

    sae_split_bf16<<<4096, 256, 0, stream>>>(x, x_hi, x_lo, N_ROWS * D_MODEL);
    sae_split_bf16<<<4096, 256, 0, stream>>>(W_enc, we_hi, we_lo, D_SAE * D_MODEL);

    dim3 tgrid(D_SAE / 32, D_MODEL / 32);
    sae_transpose_wdec<<<tgrid, dim3(32, 8), 0, stream>>>(W_dec, WdT);

    dim3 egrid(D_SAE / 128, N_ROWS / 64);
    sae_encode_wmma<<<egrid, 256, 0, stream>>>(x_hi, x_lo, we_hi, we_lo, b_enc,
                                               latents, sparse);

    const size_t topk_lds = (size_t)(D_SAE + 512) * sizeof(float); // 66 KB < 320 KB WGP LDS
    sae_topk<<<N_ROWS, 256, topk_lds, stream>>>(latents, sparse, tk_vals, tk_idx);

    sae_decode<<<N_ROWS, 256, 0, stream>>>(tk_vals, tk_idx, WdT, b_dec, rec);
}
